// ModularRIN_75892072121002
// MI455X (gfx1250) — matrix operand order repaired and checked by reading, not compiled, not run on hardware
//
#include <hip/hip_runtime.h>

typedef __attribute__((ext_vector_type(16))) _Float16 v16h;
typedef __attribute__((ext_vector_type(8)))  _Float16 v8h;
typedef __attribute__((ext_vector_type(8)))  float    v8f;

#define LUTN   4096
#define TWOPIf 6.2831853071795864f
#define INV2PI 0.15915494309189535f
#define PHIf   1.6180339887498949f

#define Bsz 2048
#define Ssz 512
#define Vsz 113
#define VP  128         // V padded to WMMA N multiple
#define Dsz 48
#define Nsz 96
#define BM  16          // batch rows per block == one WMMA M tile
#define KP  64          // D padded to multiple of 32 for WMMA K
#define NTHREADS 128    // 4 waves
#define NWAVES 4

// ---- LUT sin/cos with linear interp, tables in LDS ----
__device__ __forceinline__ void lut_sincos(float th, const float* st, const float* ct,
                                           float& s, float& c) {
    float r   = th - floorf(th * INV2PI) * TWOPIf;     // mod into [0, 2pi)
    float p   = r * ((float)LUTN / TWOPIf);
    float fi  = floorf(p);
    float fr  = p - fi;
    int i0 = ((int)fi) & (LUTN - 1);
    int i1 = (i0 + 1) & (LUTN - 1);
    float om = 1.0f - fr;
    s = st[i0] * om + st[i1] * fr;
    c = ct[i0] * om + ct[i1] * fr;
}

// A fragment: 16x32 f16 tile (lanes 0-15: M=lane, K k0..k0+15;
// lanes 16-31: M=lane-16, K k0+16..k0+31), row-major LDS source.
__device__ __forceinline__ v16h load_a_frag(const _Float16* base, int rowStride,
                                            int m0, int k0, int lane) {
    int m  = m0 + (lane & 15);
    int kb = k0 + ((lane >> 4) << 4);
    const v8h* p0 = (const v8h*)(base + m * rowStride + kb);
    const v8h* p1 = (const v8h*)(base + m * rowStride + kb + 8);
    v8h lo = *p0;
    v8h hi = *p1;
    v16h a;
#pragma unroll
    for (int i = 0; i < 8; ++i) { a[i] = lo[i]; a[i + 8] = hi[i]; }
    return a;
}

// B fragment: 32x16 f16 tile; lane holds column n0+(lane&15), contiguous 16 K
// values (lanes 0-15: K k0.., lanes 16-31: K k0+16..). Source LDS is [col][k].
__device__ __forceinline__ v16h load_b_frag(const _Float16* base, int rowStride,
                                            int n0, int k0, int lane) {
    int n  = n0 + (lane & 15);
    int kb = k0 + ((lane >> 4) << 4);
    return *(const v16h*)(base + n * rowStride + kb);
}

__global__ __launch_bounds__(NTHREADS) void modular_rin_scan_kernel(
    const int*   __restrict__ input_ids,   // [B, S]
    const float* __restrict__ emb,         // [V, 2D]
    const float* __restrict__ W0p, const float* __restrict__ b0p,
    const float* __restrict__ pr0, const float* __restrict__ pi0,
    const float* __restrict__ W1p, const float* __restrict__ b1p,
    const float* __restrict__ pr1, const float* __restrict__ pi1,
    const float* __restrict__ out_w,       // [V, D]
    float*       __restrict__ out)         // [B, V]
{
    __shared__ float    s_tab[LUTN];
    __shared__ float    c_tab[LUTN];
    __shared__ _Float16 Wh[2][Nsz][KP];        // W as [layer][n][k], k padded
    __shared__ _Float16 Ph[2][2][Dsz][Nsz];    // [layer][re/im][d][n]
    __shared__ _Float16 OWh[VP][KP];           // out_w as [v][k], padded
    __shared__ float    biasL[2][Nsz];
    __shared__ _Float16 hA[BM][KP];            // f16 h (A operand), k padded
    __shared__ _Float16 cB[BM][Nsz];           // cos activations
    __shared__ _Float16 sB[BM][Nsz];           // sin activations
    __shared__ float    hr[BM][Dsz];           // recurrence state: cos part
    __shared__ float    hi_[BM][Dsz];          // recurrence state: sin part
    __shared__ float    hcur[BM][Dsz];         // current h (f32)
    __shared__ int      ids2[2][BM];           // double-buffered token ids

    const int tid  = threadIdx.x;
    const int lane = tid & 31;
    const int wave = tid >> 5;
    const int b0   = blockIdx.x * BM;

    // ---- one-time init: LUT tables, f16 weights, zero state ----
    for (int i = tid; i < LUTN; i += NTHREADS) {
        float ang = (float)i * (TWOPIf / (float)LUTN);
        s_tab[i] = __sinf(ang);
        c_tab[i] = __cosf(ang);
    }
    for (int j = tid; j < Nsz * KP; j += NTHREADS) {
        int n = j / KP, k = j % KP;
        Wh[0][n][k] = (k < Dsz) ? (_Float16)W0p[n * Dsz + k] : (_Float16)0.0f;
        Wh[1][n][k] = (k < Dsz) ? (_Float16)W1p[n * Dsz + k] : (_Float16)0.0f;
    }
    for (int j = tid; j < Dsz * Nsz; j += NTHREADS) {
        int d = j / Nsz, n = j % Nsz;
        Ph[0][0][d][n] = (_Float16)pr0[j];
        Ph[0][1][d][n] = (_Float16)pi0[j];
        Ph[1][0][d][n] = (_Float16)pr1[j];
        Ph[1][1][d][n] = (_Float16)pi1[j];
    }
    for (int j = tid; j < VP * KP; j += NTHREADS) {
        int v = j / KP, k = j % KP;
        OWh[v][k] = (v < Vsz && k < Dsz) ? (_Float16)out_w[v * Dsz + k]
                                         : (_Float16)0.0f;
    }
    for (int j = tid; j < Nsz; j += NTHREADS) {
        biasL[0][j] = b0p[j];
        biasL[1][j] = b1p[j];
    }
    for (int j = tid; j < BM * KP; j += NTHREADS) ((_Float16*)hA)[j] = (_Float16)0.0f;
    for (int j = tid; j < BM * Dsz; j += NTHREADS) {
        ((float*)hr)[j] = 0.0f; ((float*)hi_)[j] = 0.0f; ((float*)hcur)[j] = 0.0f;
    }
    if (tid < BM) ids2[0][tid] = input_ids[(size_t)(b0 + tid) * Ssz];
    __syncthreads();

    // ---- sequential scan over S ----
    for (int s = 0; s < Ssz; ++s) {
        const float t_val = (float)s * PHIf;
        const int*  idcur = ids2[s & 1];

        // prefetch next step's token ids into the other buffer (overlaps with
        // this step's compute; consumed only after this step's layer barriers)
        if (tid < BM && (s + 1) < Ssz)
            ids2[(s + 1) & 1][tid] = input_ids[(size_t)(b0 + tid) * Ssz + s + 1];

        // recurrence: theta = (hr+hi)/wavelength + b_t + t; (hi,hr) = (sin,cos)
        for (int j = tid; j < BM * Dsz; j += NTHREADS) {
            int r = j / Dsz, d = j % Dsz;
            const float* e = emb + (size_t)idcur[r] * (2 * Dsz);
            float w  = e[d];
            float bt = e[Dsz + d];
            float wavelength = 1.0f + fabsf(w);
            float th = (hr[r][d] + hi_[r][d]) / wavelength + bt + t_val;
            float si, co;
            lut_sincos(th, s_tab, c_tab, si, co);
            hr[r][d]  = co;
            hi_[r][d] = si;
            float h = co + si;
            hcur[r][d] = h;
            hA[r][d]   = (_Float16)h;
        }
        __syncthreads();

        // ---- 2 residual layers ----
        for (int l = 0; l < 2; ++l) {
            // theta = h @ W^T + bias + t ; then LUT sin/cos -> sB/cB (f16)
            for (int t = wave; t < 6; t += NWAVES) {    // N/16 = 6 tiles, M tile = 1
                v8f acc = {};
#pragma unroll
                for (int k0 = 0; k0 < KP; k0 += 32) {
                    v16h a = load_a_frag(&hA[0][0], KP, 0, k0, lane);
                    v16h b = load_b_frag(&Wh[l][0][0], KP, t * 16, k0, lane);
                    acc = __builtin_amdgcn_wmma_f32_16x16x32_f16(
                        false, a, false, b, (short)0, acc, false, false);
                }
                int rbase = (lane >> 4) << 3;
                int col   = t * 16 + (lane & 15);
                float bias_t = biasL[l][col] + t_val;
#pragma unroll
                for (int j = 0; j < 8; ++j) {
                    float th = acc[j] + bias_t;
                    float si, co;
                    lut_sincos(th, s_tab, c_tab, si, co);
                    sB[rbase + j][col] = (_Float16)si;
                    cB[rbase + j][col] = (_Float16)co;
                }
            }
            __syncthreads();

            // out = c @ pr^T + s @ pi^T ; h += silu(out)
            for (int t = wave; t < 3; t += NWAVES) {    // D/16 = 3 tiles
                v8f acc = {};
#pragma unroll
                for (int k0 = 0; k0 < Nsz; k0 += 32) {
                    v16h ac = load_a_frag(&cB[0][0], Nsz, 0, k0, lane);
                    v16h bp = load_b_frag(&Ph[l][0][0][0], Nsz, t * 16, k0, lane);
                    acc = __builtin_amdgcn_wmma_f32_16x16x32_f16(
                        false, ac, false, bp, (short)0, acc, false, false);
                    v16h as = load_a_frag(&sB[0][0], Nsz, 0, k0, lane);
                    v16h bi = load_b_frag(&Ph[l][1][0][0], Nsz, t * 16, k0, lane);
                    acc = __builtin_amdgcn_wmma_f32_16x16x32_f16(
                        false, as, false, bi, (short)0, acc, false, false);
                }
                int rbase = (lane >> 4) << 3;
                int col   = t * 16 + (lane & 15);
#pragma unroll
                for (int j = 0; j < 8; ++j) {
                    float o   = acc[j];
                    float sil = o / (1.0f + __expf(-o));     // silu
                    float hn  = hcur[rbase + j][col] + sil;
                    hcur[rbase + j][col] = hn;
                    hA[rbase + j][col]   = (_Float16)hn;
                }
            }
            __syncthreads();
        }
    }

    // ---- final projection via WMMA: out[b, v] = sum_d h[b,d] * out_w[v,d] ----
    for (int t = wave; t < VP / 16; t += NWAVES) {      // 8 N-tiles over 4 waves
        v8f acc = {};
#pragma unroll
        for (int k0 = 0; k0 < KP; k0 += 32) {
            v16h a = load_a_frag(&hA[0][0], KP, 0, k0, lane);
            v16h b = load_b_frag(&OWh[0][0], KP, t * 16, k0, lane);
            acc = __builtin_amdgcn_wmma_f32_16x16x32_f16(
                false, a, false, b, (short)0, acc, false, false);
        }
        int rbase = (lane >> 4) << 3;
        int col   = t * 16 + (lane & 15);
        if (col < Vsz) {
#pragma unroll
            for (int j = 0; j < 8; ++j)
                out[(size_t)(b0 + rbase + j) * Vsz + col] = acc[j];
        }
    }
}

extern "C" void kernel_launch(void* const* d_in, const int* in_sizes, int n_in,
                              void* d_out, int out_size, void* d_ws, size_t ws_size,
                              hipStream_t stream) {
    (void)in_sizes; (void)n_in; (void)d_ws; (void)ws_size; (void)out_size;
    const int*   input_ids = (const int*)  d_in[0];
    const float* emb       = (const float*)d_in[1];
    const float* W0        = (const float*)d_in[2];
    const float* b0        = (const float*)d_in[3];
    const float* pr0       = (const float*)d_in[4];
    const float* pi0       = (const float*)d_in[5];
    const float* W1        = (const float*)d_in[6];
    const float* b1        = (const float*)d_in[7];
    const float* pr1       = (const float*)d_in[8];
    const float* pi1       = (const float*)d_in[9];
    const float* out_w     = (const float*)d_in[10];
    float*       out       = (float*)d_out;

    dim3 grid(Bsz / BM);      // 128 blocks: one 16-row WMMA M-tile each
    dim3 block(NTHREADS);     // 4 waves
    modular_rin_scan_kernel<<<grid, block, 0, stream>>>(
        input_ids, emb, W0, b0, pr0, pi0, W1, b1, pr1, pi1, out_w, out);
}
